// GraphNetworkSimulator_36386962931762
// MI455X (gfx1250) — matrix operand order repaired and checked by reading, not compiled, not run on hardware
//
#include <hip/hip_runtime.h>

// ---------------------------------------------------------------------------
// Fused GraphNetworkSimulator for gfx1250 (MI455X), wave32 + WMMA f16.
// One wave = 16 graphs (80 node rows = 5 WMMA M-tiles, 64 edge rows = 4 tiles).
// All 19 MLP layers run as v_wmma_f32_16x16x32_f16 with inline-zero C;
// biases ride in the K-padding as a weight row against a constant-1.0 A column
// held in LDS (so A-fragments assemble via two direct ds_load_b128, no movs).
// Residual trunk (h_n, h_e) lives in f32 C-fragments in registers; LDS holds
// only the f16 A-operand copies of activations.
// ---------------------------------------------------------------------------

typedef _Float16 half16 __attribute__((ext_vector_type(16)));
typedef _Float16 half8  __attribute__((ext_vector_type(8)));
typedef float    float8 __attribute__((ext_vector_type(8)));

#define N_GRAPHS   131072
#define WAVES      4
#define BLOCKDIM   (WAVES * 32)
#define GPW        16              // graphs per wave
#define GPB        (GPW * WAVES)   // graphs per block
#define NBLOCKS    (N_GRAPHS / GPB)

#define DTc        0.01f
#define ALPHAc     2.0f
#define MASSc      1500.0f
#define NOISESTDc  0.0003f

// 19 GEMM layers, in order:
// 0 enc_node L1, 1 enc_node L2, 2 enc_edge L1, 3 enc_edge L2,
// 4..15 proc s={0,1,2}: edgeL1, edgeL2, nodeL1, nodeL2,
// 16 dec L1, 17 dec L2, 18 dec L3 (dout=1)
__device__ __constant__ int d_TOFF[19] = {0,1,2,3, 4,6,7,9, 10,12,13,15, 16,18,19,21, 22,23,24};

struct PackArgs {
  const float* W[19];
  const float* Bv[19];
};

// Pack weights into WMMA B-operand lane-major tiles (32x16 K-tiles of f16):
// dst[tile*512 + lane*16 + i] with lane = h*16 + n, klocal = (i>=8)*16 + h*8 + i%8.
// The bias is embedded as an extra "weight row" at k = 16 (din<=16) or k = 48
// (din==48); the main kernel feeds a constant-1.0 A column at that k.
__global__ void gns_pack(PackArgs args, _Float16* __restrict__ wpack) {
  const int DIN[19]  = {10,16,1,16, 48,16,48,16, 48,16,48,16, 48,16,48,16, 16,16,16};
  const int DOUT[19] = {16,16,16,16, 16,16,16,16, 16,16,16,16, 16,16,16,16, 16,16,1};
  int tid = threadIdx.x + blockIdx.x * blockDim.x;
  int stride = blockDim.x * gridDim.x;
  for (int idx = tid; idx < 25 * 512; idx += stride) {
    int t = idx >> 9;
    int e = idx & 511;
    int l = 0;
    #pragma unroll
    for (int i = 0; i < 19; i++) if (t >= d_TOFF[i]) l = i;
    int kt   = t - d_TOFF[l];
    int lane = e >> 4;
    int i    = e & 15;
    int k    = kt * 32 + ((i >> 3) << 4) + ((lane >> 4) << 3) + (i & 7);   // klocal = (i>=8)*16 + (lane>=16)*8 + i%8
    int n    = lane & 15;
    int bk   = (DIN[l] == 48) ? 48 : 16;           // bias row position
    float v  = 0.0f;
    if (n < DOUT[l]) {
      if (k < DIN[l])      v = args.W[l][k * DOUT[l] + n];
      else if (k == bk)    v = args.Bv[l][n];
    }
    wpack[idx] = (_Float16)v;
  }
}

// Per-wave LDS regions (units of _Float16)
#define R_RAWN 0      // [80][16] raw node feats (f16); reused by decoder hidden
#define R_RAWE 1280   // [64][16] raw edge feats
#define R_HN   2304   // [80][16] f16 copy of node latents (A-operand source)
#define R_HE   3584   // [64][16] f16 copy of edge latents
#define R_NE   4608   // [64][16] new_e of current proc step
#define R_TMP  5632   // [80][16] hidden-layer scratch
#define R_CTRL 6912   // 80 floats (160 halfs) decoded controls
#define R_ZROW 7072   // 16 halfs of zeros (true-zero gathers)
#define R_ONE  7088   // 16 halfs of 1.0 (bias column)
#define WSTRIDE 7104

__device__ __forceinline__ float8 wmma_f16(half16 a, half16 b, float8 c) {
  return __builtin_amdgcn_wmma_f32_16x16x32_f16(false, a, false, b, (short)0, c,
                                                false, false);
}

__device__ __forceinline__ float relu1(float v) {
  // guaranteed single instruction (compiler's fmax lowering adds a canonicalize)
  float r;
  asm("v_max_num_f32_e32 %0, 0, %1" : "=v"(r) : "v"(v));
  return r;
}

__launch_bounds__(BLOCKDIM)
__global__ void gns_main(const float* __restrict__ nodes,
                         const float* __restrict__ edges,
                         const float* __restrict__ noise,
                         const _Float16* __restrict__ wpack,
                         float* __restrict__ out) {
  __shared__ _Float16 lds[WAVES * WSTRIDE];
  const int tid  = threadIdx.x;
  const int wave = tid >> 5;
  const int lane = tid & 31;
  const int mrow = lane & 15;        // row within 16-row M-tile (A layout)
  const int hi   = lane >> 4;        // lane half select
  const int coff = hi << 3;          // column offset (in halfs) for A builds
  const int col  = mrow;             // C-matrix column for this lane
  _Float16* L = lds + wave * WSTRIDE;
  const int gbase = (blockIdx.x * WAVES + wave) * GPW;

  // A-operand build: lane<16 -> K {0..7 from lo, 16..23 from hiArr},
  //                  lane>=16 -> K {8..15 from lo, 24..31 from hiArr}
  // Both halves load straight into the A register tuple (ds_load_b128 x2).
  auto mkA = [&](const _Float16* rowLo, const _Float16* rowHi) -> half16 {
    union { half16 v; half8 h[2]; } u;
    u.h[0] = *(const half8*)(rowLo + coff);
    u.h[1] = *(const half8*)(rowHi + coff);
    return u.v;
  };
  auto loadB = [&](int tile) -> half16 {
    return *(const half16*)(wpack + (size_t)tile * 512 + lane * 16);
  };
  // C tile -> LDS f16 (VGPR j holds M = hi*8 + j, N = col)
  auto storeH = [&](float8 acc, _Float16* tileBase, bool relu) {
    #pragma unroll
    for (int j = 0; j < 8; j++) {
      float v = acc[j];
      if (relu) v = relu1(v);
      tileBase[(hi * 8 + j) * 16 + col] = (_Float16)v;
    }
  };
  auto zeroC = [&]() -> float8 {
    float8 c = {0.0f, 0.0f, 0.0f, 0.0f, 0.0f, 0.0f, 0.0f, 0.0f};
    return c;
  };

  // ---- stage raw inputs (vectorized f16 packing, ds_store_b128) ----
  if (lane < 16) {
    L[R_ZROW + lane] = (_Float16)0.0f;
    L[R_ONE  + lane] = (_Float16)1.0f;
  }
  for (int r = lane; r < 80; r += 32) {
    const float2* s2 = (const float2*)(nodes + (size_t)(gbase + r / 5) * 50 +
                                       (size_t)(r % 5) * 10);   // 8B aligned
    float2 f0 = s2[0], f1 = s2[1], f2 = s2[2], f3 = s2[3], f4 = s2[4];
    union { half8 v; _Float16 e[8]; } lo, hh;
    lo.e[0] = (_Float16)f0.x; lo.e[1] = (_Float16)f0.y;
    lo.e[2] = (_Float16)f1.x; lo.e[3] = (_Float16)f1.y;
    lo.e[4] = (_Float16)f2.x; lo.e[5] = (_Float16)f2.y;
    lo.e[6] = (_Float16)f3.x; lo.e[7] = (_Float16)f3.y;
    hh.e[0] = (_Float16)f4.x; hh.e[1] = (_Float16)f4.y;
    #pragma unroll
    for (int c = 2; c < 8; c++) hh.e[c] = (_Float16)0.0f;
    _Float16* dst = L + R_RAWN + r * 16;
    *(half8*)dst = lo.v;
    *(half8*)(dst + 8) = hh.v;
  }
  for (int r = lane; r < 64; r += 32) {
    float v = edges[(size_t)(gbase + (r >> 2)) * 4 + (r & 3)];
    union { half8 v; _Float16 e[8]; } u, z;
    u.e[0] = (_Float16)v;
    #pragma unroll
    for (int c = 1; c < 8; c++) u.e[c] = (_Float16)0.0f;
    #pragma unroll
    for (int c = 0; c < 8; c++) z.e[c] = (_Float16)0.0f;
    _Float16* dst = L + R_RAWE + r * 16;
    *(half8*)dst = u.v;
    *(half8*)(dst + 8) = z.v;
  }
  const _Float16* ZR = L + R_ZROW;   // true zeros (graph-boundary gathers)
  const _Float16* ON = L + R_ONE;    // 1.0 column (bias row activation)

  float8 hn[5];   // node latents, f32 C-fragments (residual trunk in registers)
  float8 he[4];   // edge latents

  // ---- encoder ----
  {
    half16 b0 = loadB(0);
    #pragma unroll
    for (int t = 0; t < 5; t++) {
      float8 acc = wmma_f16(mkA(L + R_RAWN + (t * 16 + mrow) * 16, ON), b0, zeroC());
      storeH(acc, L + R_TMP + t * 256, true);
    }
    half16 b1 = loadB(1);
    #pragma unroll
    for (int t = 0; t < 5; t++) {
      float8 acc = wmma_f16(mkA(L + R_TMP + (t * 16 + mrow) * 16, ON), b1, zeroC());
      hn[t] = acc;
      storeH(acc, L + R_HN + t * 256, false);
    }
    half16 b2 = loadB(2);
    #pragma unroll
    for (int t = 0; t < 4; t++) {
      float8 acc = wmma_f16(mkA(L + R_RAWE + (t * 16 + mrow) * 16, ON), b2, zeroC());
      storeH(acc, L + R_TMP + t * 256, true);
    }
    half16 b3 = loadB(3);
    #pragma unroll
    for (int t = 0; t < 4; t++) {
      float8 acc = wmma_f16(mkA(L + R_TMP + (t * 16 + mrow) * 16, ON), b3, zeroC());
      he[t] = acc;
      storeH(acc, L + R_HE + t * 256, false);
    }
  }

  // ---- processor: 3 message-passing steps (fully unrolled) ----
  #pragma unroll
  for (int s = 0; s < 3; s++) {
    const int tE1 = 4 + 6 * s;       // edge L1 tiles: tE1, tE1+1
    const int tE2 = 6 + 6 * s;       // edge L2 tile
    const int tN1 = 7 + 6 * s;       // node L1 tiles: tN1, tN1+1
    const int tN2 = 9 + 6 * s;       // node L2 tile

    // edge MLP: in = [h_e | h_n[snd] | h_n[rcv] | bias] (two K=32 WMMAs)
    half16 bE1a = loadB(tE1);
    half16 bE1b = loadB(tE1 + 1);
    #pragma unroll
    for (int t = 0; t < 4; t++) {
      int re = t * 16 + mrow;
      int g = re >> 2, e = re & 3;
      float8 acc = wmma_f16(mkA(L + R_HE + re * 16, L + R_HN + (g * 5 + e) * 16),
                            bE1a, zeroC());
      acc = wmma_f16(mkA(L + R_HN + (g * 5 + e + 1) * 16, ON), bE1b, acc);
      storeH(acc, L + R_TMP + t * 256, true);
    }
    half16 bE2 = loadB(tE2);
    #pragma unroll
    for (int t = 0; t < 4; t++) {
      float8 acc = wmma_f16(mkA(L + R_TMP + (t * 16 + mrow) * 16, ON), bE2, zeroC());
      storeH(acc, L + R_NE + t * 256, false);   // new_e (f16 for node gathers)
      he[t] += acc;                             // residual in f32 registers
      storeH(he[t], L + R_HE + t * 256, false); // refresh f16 copy
    }

    // node MLP: in = [h_n | sent | recv | bias]; chain graph => pure gathers
    half16 bN1a = loadB(tN1);
    half16 bN1b = loadB(tN1 + 1);
    #pragma unroll
    for (int t = 0; t < 5; t++) {
      int rn = t * 16 + mrow;
      int g = rn / 5, n = rn % 5;
      const _Float16* sent = (n < 4)  ? (L + R_NE + (g * 4 + n) * 16)     : ZR;
      const _Float16* recv = (n >= 1) ? (L + R_NE + (g * 4 + n - 1) * 16) : ZR;
      float8 acc = wmma_f16(mkA(L + R_HN + rn * 16, sent), bN1a, zeroC());
      acc = wmma_f16(mkA(recv, ON), bN1b, acc);
      storeH(acc, L + R_TMP + t * 256, true);
    }
    half16 bN2 = loadB(tN2);
    #pragma unroll
    for (int t = 0; t < 5; t++) {
      float8 acc = wmma_f16(mkA(L + R_TMP + (t * 16 + mrow) * 16, ON), bN2, zeroC());
      hn[t] += acc;                             // residual in f32 registers
      storeH(hn[t], L + R_HN + t * 256, false); // refresh f16 copy
    }
  }

  // ---- decoder ----
  float* ctrlF = (float*)(L + R_CTRL);
  {
    half16 bD1 = loadB(22);
    #pragma unroll
    for (int t = 0; t < 5; t++) {
      float8 acc = wmma_f16(mkA(L + R_HN + (t * 16 + mrow) * 16, ON), bD1, zeroC());
      storeH(acc, L + R_TMP + t * 256, true);
    }
    half16 bD2 = loadB(23);
    #pragma unroll
    for (int t = 0; t < 5; t++) {
      float8 acc = wmma_f16(mkA(L + R_TMP + (t * 16 + mrow) * 16, ON), bD2, zeroC());
      storeH(acc, L + R_RAWN + t * 256, true);  // reuse raw-node region
    }
    half16 bD3 = loadB(24);
    #pragma unroll
    for (int t = 0; t < 5; t++) {
      float8 acc = wmma_f16(mkA(L + R_RAWN + (t * 16 + mrow) * 16, ON), bD3, zeroC());
      if (col == 0) {                            // only column 0 is the control
        #pragma unroll
        for (int j = 0; j < 8; j++) ctrlF[t * 16 + hi * 8 + j] = acc[j];
      }
    }
  }

  // ---- platoon dynamics (closed form; same-wave LDS ops are in-order) ----
  if (lane < 16) {
    const int gg = gbase + lane;
    float p[5], v[5], c[5], nz[5];
    #pragma unroll
    for (int n = 0; n < 5; n++) {
      const float* sp = nodes + (size_t)(gg * 5 + n) * 10;
      p[n]  = sp[0];
      v[n]  = sp[1];
      c[n]  = ctrlF[lane * 5 + n];
      nz[n] = noise[(size_t)gg * 5 + n];
    }
    const float dtm = DTc / MASSc;
    #pragma unroll
    for (int n = 0; n < 5; n++) {
      float np = p[n] + DTc * v[n];
      float nv = v[n] + dtm * c[n] + dtm * NOISESTDc * nz[n];
      if (n >= 1) nv += (ALPHAc * DTc / MASSc) * (p[n] - p[n - 1]);
      float* o = out + (size_t)(gg * 5 + n) * 3;
      o[0] = np; o[1] = nv; o[2] = c[n];
    }
  }
}

extern "C" void kernel_launch(void* const* d_in, const int* in_sizes, int n_in,
                              void* d_out, int out_size, void* d_ws, size_t ws_size,
                              hipStream_t stream) {
  (void)in_sizes; (void)n_in; (void)out_size; (void)ws_size;
  const float* nodes = (const float*)d_in[0];
  const float* edges = (const float*)d_in[1];
  const float* noise = (const float*)d_in[2];

  // Params flattened in JAX pytree order (sorted dict keys):
  // dec_node[3..8], enc_edge[9..12], enc_node[13..16], proc[17..40]
  static const int WI[19] = {13,15, 9,11, 17,19,21,23, 25,27,29,31, 33,35,37,39, 3,5,7};
  static const int BI[19] = {14,16, 10,12, 18,20,22,24, 26,28,30,32, 34,36,38,40, 4,6,8};
  PackArgs pa;
  for (int l = 0; l < 19; l++) {
    pa.W[l]  = (const float*)d_in[WI[l]];
    pa.Bv[l] = (const float*)d_in[BI[l]];
  }

  _Float16* wpack = (_Float16*)d_ws;   // 25 tiles * 1KB (biases embedded)

  hipLaunchKernelGGL(gns_pack, dim3(52), dim3(256), 0, stream, pa, wpack);
  hipLaunchKernelGGL(gns_main, dim3(NBLOCKS), dim3(BLOCKDIM), 0, stream,
                     nodes, edges, noise, wpack, (float*)d_out);
}
